// MultiHeadAttention_58712202936854
// MI455X (gfx1250) — matrix operand order repaired and checked by reading, not compiled, not run on hardware
//
#include <hip/hip_runtime.h>

typedef __attribute__((ext_vector_type(16))) __bf16 v16bf;
typedef __attribute__((ext_vector_type(8)))  __bf16 v8bf;
typedef __attribute__((ext_vector_type(8)))  float  v8f;
typedef unsigned int u32;
typedef u32 u32x4 __attribute__((ext_vector_type(4)));
typedef u32 u32x8 __attribute__((ext_vector_type(8)));

#define DIM   512
#define NHEAD 8
#define HD    64
#define NB    4096
#define BATCH 32
#define SEQ   128
#define NTOK  4096   // BATCH*SEQ

// LDS row pitches (bf16 elements) after TDM hardware padding (16B pad per row)
#define KV_PITCH 72    // 128B data + 16B pad
#define W_PITCH  520   // 1024B data + 16B pad

#define WMMA_BF16(a, b, c) \
  __builtin_amdgcn_wmma_f32_16x16x32_bf16(false, (a), false, (b), (short)0, (c), false, false)

// ---------------------------------------------------------------- TDM issue
// 2D tile load Global->LDS via Tensor Data Mover (D# groups 0+1, VADDR2/3=NULL).
// All dims/strides in data_size(=8B) units. Optional LDS padding.
// pad_int_code: pad every 2^(code+1) DWORDs; pad_amt_code: (code+1) DWORDs pad.
static __device__ __forceinline__ void tdm_load_2d(u32 lds_off, const void* gptr,
                                                   u32 tensor_d0, u32 tensor_d1,
                                                   u32 tile_d0, u32 tile_d1, u32 stride0,
                                                   u32 pad_en, u32 pad_int_code, u32 pad_amt_code) {
  unsigned long long ga = (unsigned long long)gptr;
  u32x4 g0;
  g0[0] = 1u;                                              // count=1 (valid), user mode
  g0[1] = lds_off;                                         // lds_addr
  g0[2] = (u32)ga;                                         // global_addr[31:0]
  g0[3] = (u32)((ga >> 32) & 0x01ffffffu) | 0x80000000u;   // addr[56:32] | type=2
  u32x8 g1;
  g1[0] = (3u << 16) | (pad_en << 20) | (pad_int_code << 22) | (pad_amt_code << 25);
  g1[1] = (tensor_d0 & 0xffffu) << 16;                     // tensor_dim0[15:0]
  g1[2] = ((tensor_d0 >> 16) & 0xffffu) | ((tensor_d1 & 0xffffu) << 16);
  g1[3] = ((tensor_d1 >> 16) & 0xffffu) | ((tile_d0 & 0xffffu) << 16);
  g1[4] = (tile_d1 & 0xffffu);                             // tile_dim1, tile_dim2=0
  g1[5] = stride0;                                         // tensor_dim0_stride[31:0]
  g1[6] = 0u;
  g1[7] = 0u;
  asm volatile("tensor_load_to_lds %0, %1" :: "s"(g0), "s"(g1) : "memory");
}

// A-matrix fragment (contiguous k order, matching B-frag loads):
// lanes 0-15  (sub=0): row=lane,    K = [k0..k0+15]
// lanes 16-31 (sub=1): row=lane-16, K = [k0+16..k0+31]
static __device__ __forceinline__ v16bf load_a_frag(const __bf16* rowbase, int k0, int sub) {
  const __bf16* p = rowbase + k0 + sub * 16;
  v8bf lo = *(const v8bf*)(p);
  v8bf hi = *(const v8bf*)(p + 8);
  v16bf r;
#pragma unroll
  for (int i = 0; i < 8; i++) { r[i] = lo[i]; r[i + 8] = hi[i]; }
  return r;
}

// ---------------------------------------------------------------- cvt f32->bf16
__global__ __launch_bounds__(256) void cvt_f32_bf16(const float* __restrict__ in,
                                                    __bf16* __restrict__ out, int n) {
  int i = blockIdx.x * 256 + threadIdx.x;
  if (i < n) out[i] = (__bf16)in[i];
}

// ---------------------------------------------------------------- C = A * W^T
// A: M x 512 bf16 row-major, W: 512 x 512 bf16 row-major (torch y = x @ W.T)
// block = 256 (8 waves). Wave w: rows [bm*128+w*16, +16), cols [bn*64, +64).
// The 64x512 W slab is staged once into LDS by TDM (shared by all 8 waves).
__global__ __launch_bounds__(256) void gemm_bf16_nt(const __bf16* __restrict__ A,
                                                    const __bf16* __restrict__ W,
                                                    float* __restrict__ C) {
  __shared__ __align__(16) __bf16 ldsW[64 * W_PITCH];   // 66,560 B
  const int wave = threadIdx.x >> 5;
  const int lane = threadIdx.x & 31;
  const int sub  = lane >> 4;
  const int l15  = lane & 15;
  const int m0 = blockIdx.x * 128 + wave * 16;
  const int n0 = blockIdx.y * 64;

  if (wave == 0) {
    // 64 KB contiguous slab (rows n0..n0+63 of W); pad 16B every 1024B row.
    tdm_load_2d((u32)(size_t)&ldsW[0], W + (size_t)n0 * DIM,
                /*tensor_d0=*/8192, /*tensor_d1=*/1,
                /*tile_d0=*/8192, /*tile_d1=*/1, /*stride0=*/8192,
                /*pad_en=*/1, /*pad_int=*/7, /*pad_amt=*/3);
    __builtin_amdgcn_s_wait_tensorcnt(0);
  }
  __syncthreads();

  const __bf16* arow = A + (size_t)(m0 + l15) * DIM;
  const __bf16* wrow = &ldsW[(size_t)l15 * W_PITCH + sub * 16];
  v8f acc[4] = {};
#pragma unroll 1
  for (int k0 = 0; k0 < DIM; k0 += 32) {
    v16bf a = load_a_frag(arow, k0, sub);
#pragma unroll
    for (int t = 0; t < 4; t++) {
      v16bf b = *(const v16bf*)(wrow + (size_t)t * 16 * W_PITCH + k0);
      acc[t] = WMMA_BF16(a, b, acc[t]);
    }
  }
#pragma unroll
  for (int t = 0; t < 4; t++)
#pragma unroll
    for (int j = 0; j < 8; j++)
      C[(size_t)(m0 + j + sub * 8) * DIM + n0 + t * 16 + l15] = acc[t][j];
}

// ---------------------------------------------------------------- head split + LN
// mode 0: q -> LN(qn)*scale, layout ((b*H+h)*SEQ+s)*HD   (row = b*SEQ+s)
// mode 1: k -> LN(kn),       layout (h*NB+n)*HD
// mode 2: v -> no LN,        layout (h*HD+hd)*NB + n     (transposed)
__global__ __launch_bounds__(256) void headsplit(const float* __restrict__ src,
                                                 __bf16* __restrict__ dst,
                                                 const float* __restrict__ g,
                                                 const float* __restrict__ bb,
                                                 int mode) {
  const int task = blockIdx.x * 8 + (threadIdx.x >> 5);  // 4096*8 tasks
  const int lane = threadIdx.x & 31;
  const int row = task >> 3;
  const int h   = task & 7;
  const float* p = src + (size_t)row * DIM + h * HD;
  float e0 = p[lane], e1 = p[lane + 32];
  float out0 = e0, out1 = e1;
  if (mode < 2) {
    float s = e0 + e1;
#pragma unroll
    for (int m = 16; m >= 1; m >>= 1) s += __shfl_xor(s, m, 32);
    float mean = s * (1.0f / 64.0f);
    float d0 = e0 - mean, d1 = e1 - mean;
    float v = d0 * d0 + d1 * d1;
#pragma unroll
    for (int m = 16; m >= 1; m >>= 1) v += __shfl_xor(v, m, 32);
    float rstd = rsqrtf(v * (1.0f / 64.0f) + 1e-5f);
    float sc = (mode == 0) ? 0.125f : 1.0f;  // HD^-0.5 folded into q
    out0 = (d0 * rstd * g[lane]      + bb[lane])      * sc;
    out1 = (d1 * rstd * g[lane + 32] + bb[lane + 32]) * sc;
  }
  if (mode == 2) {
    dst[(size_t)(h * HD + lane)      * NB + row] = (__bf16)out0;
    dst[(size_t)(h * HD + lane + 32) * NB + row] = (__bf16)out1;
  } else if (mode == 1) {
    __bf16* d = dst + (size_t)(h * NB + row) * HD;
    d[lane] = (__bf16)out0; d[lane + 32] = (__bf16)out1;
  } else {
    int b = row >> 7, s = row & 127;
    __bf16* d = dst + (size_t)((b * NHEAD + h) * SEQ + s) * HD;
    d[lane] = (__bf16)out0; d[lane + 32] = (__bf16)out1;
  }
}

// ---------------------------------------------------------------- flash attention
// grid = B*H blocks of 256 threads (8 waves); wave w handles q rows [w*16, w*16+16).
// K/V chunks (64 bank rows) are TDM-staged into double-buffered LDS, shared by
// all 8 waves (8x less L2 traffic), copy overlapped with compute.
// Q: ((b*H+h)*SEQ+s)*HD   K: (h*NB+n)*HD   VT: (h*HD+hd)*NB   Out: [b][s][h][hd] f32
__global__ __launch_bounds__(256) void attention(const __bf16* __restrict__ Q,
                                                 const __bf16* __restrict__ K,
                                                 const __bf16* __restrict__ VT,
                                                 float* __restrict__ Out) {
  __shared__ __align__(16) __bf16 ldsK[2][64 * KV_PITCH];  // 2 x 9216 B
  __shared__ __align__(16) __bf16 ldsV[2][64 * KV_PITCH];  // 2 x 9216 B
  __shared__ __align__(32) __bf16 ldsP[8][16 * 64];        // 16 KB P staging
  const int wave = threadIdx.x >> 5;
  const int lane = threadIdx.x & 31;
  const int sub  = lane >> 4;
  const int l15  = lane & 15;
  const int bh = blockIdx.x;          // b*H + h
  const int b  = bh >> 3, h = bh & 7;
  const int s0 = wave * 16;

  const __bf16* qrow  = Q  + (size_t)(bh * SEQ + s0 + l15) * HD;
  const __bf16* kbase = K  + (size_t)h * NB * HD;
  const __bf16* vbase = VT + (size_t)h * HD * NB;

  // q A-frags are loop-invariant (K=64 -> two k-steps of 32)
  v16bf aq0 = load_a_frag(qrow, 0,  sub);
  v16bf aq1 = load_a_frag(qrow, 32, sub);

  v8f oacc[4] = {};
  float mrow[8], lrow[8];
#pragma unroll
  for (int j = 0; j < 8; j++) { mrow[j] = -3.0e38f; lrow[j] = 0.0f; }
  __bf16* P = &ldsP[wave][0];

  // prologue: stage chunk 0
  if (wave == 0) {
    tdm_load_2d((u32)(size_t)&ldsK[0][0], kbase, 16, 64, 16, 64, 16, 1, 4, 3);
    tdm_load_2d((u32)(size_t)&ldsV[0][0], vbase, 16, 64, 16, 64, 1024, 1, 4, 3);
    __builtin_amdgcn_s_wait_tensorcnt(0);
  }
  __syncthreads();

#pragma unroll 1
  for (int ci = 0; ci < NB / 64; ci++) {
    const int buf = ci & 1;
    // ---- kick off next chunk's TDM copies into the other buffer
    if (wave == 0 && ci + 1 < NB / 64) {
      const int n1 = (ci + 1) * 64;
      tdm_load_2d((u32)(size_t)&ldsK[buf ^ 1][0], kbase + (size_t)n1 * HD,
                  16, 64, 16, 64, 16, 1, 4, 3);
      tdm_load_2d((u32)(size_t)&ldsV[buf ^ 1][0], vbase + n1,
                  16, 64, 16, 64, 1024, 1, 4, 3);
    }
    const __bf16* kc = &ldsK[buf][0];
    const __bf16* vc = &ldsV[buf][0];

    // ---- scores S = q @ k^T : 4 column tiles x 2 k-steps (B-frags from LDS)
    v8f st[4];
#pragma unroll
    for (int t = 0; t < 4; t++) {
      const __bf16* kr = kc + (size_t)(t * 16 + l15) * KV_PITCH + sub * 16;
      v16bf b0 = *(const v16bf*)(kr);
      v16bf b1 = *(const v16bf*)(kr + 32);
      v8f c = {};
      c = WMMA_BF16(aq0, b0, c);
      st[t] = WMMA_BF16(aq1, b1, c);
    }
    // ---- online softmax update (rows live across 16 lanes of each half)
    float p[4][8];
#pragma unroll
    for (int j = 0; j < 8; j++) {
      float mx = fmaxf(fmaxf(st[0][j], st[1][j]), fmaxf(st[2][j], st[3][j]));
#pragma unroll
      for (int m = 8; m >= 1; m >>= 1) mx = fmaxf(mx, __shfl_xor(mx, m, 32));
      float mnew  = fmaxf(mrow[j], mx);
      float alpha = __expf(mrow[j] - mnew);
      float rsum = 0.0f;
#pragma unroll
      for (int t = 0; t < 4; t++) { p[t][j] = __expf(st[t][j] - mnew); rsum += p[t][j]; }
#pragma unroll
      for (int m = 8; m >= 1; m >>= 1) rsum += __shfl_xor(rsum, m, 32);
      lrow[j] = lrow[j] * alpha + rsum;
      mrow[j] = mnew;
#pragma unroll
      for (int t = 0; t < 4; t++) oacc[t][j] *= alpha;
    }
    // ---- stage P (C-layout -> row-major LDS -> A-layout reload)
#pragma unroll
    for (int t = 0; t < 4; t++)
#pragma unroll
      for (int j = 0; j < 8; j++)
        P[(j + sub * 8) * 64 + t * 16 + l15] = (__bf16)p[t][j];
    // ---- O += P @ V : 4 hd-tiles x 2 k-steps (B-frags from LDS)
#pragma unroll
    for (int ks = 0; ks < 2; ks++) {
      v16bf ap = load_a_frag(P + l15 * 64, ks * 32, sub);
#pragma unroll
      for (int t = 0; t < 4; t++) {
        v16bf bv = *(const v16bf*)(vc + (size_t)(t * 16 + l15) * KV_PITCH + ks * 32 + sub * 16);
        oacc[t] = WMMA_BF16(ap, bv, oacc[t]);
      }
    }
    // ---- ensure next chunk landed, then publish buffers to all waves
    if (wave == 0) __builtin_amdgcn_s_wait_tensorcnt(0);
    __syncthreads();
  }
  // ---- epilogue: normalize and scatter to [b][s][h][hd]
#pragma unroll
  for (int t = 0; t < 4; t++)
#pragma unroll
    for (int j = 0; j < 8; j++) {
      int row = s0 + j + sub * 8;
      Out[(size_t)(b * SEQ + row) * DIM + h * HD + t * 16 + l15] = oacc[t][j] / lrow[j];
    }
}

// ---------------------------------------------------------------- final LN over D=512 -> bf16
__global__ __launch_bounds__(256) void row_ln512(const float* __restrict__ X,
                                                 const float* __restrict__ g,
                                                 const float* __restrict__ bb,
                                                 __bf16* __restrict__ Y) {
  const int row  = blockIdx.x * 8 + (threadIdx.x >> 5);
  const int lane = threadIdx.x & 31;
  const float* x = X + (size_t)row * DIM;
  float e[16], s = 0.0f;
#pragma unroll
  for (int i = 0; i < 16; i++) { e[i] = x[lane + i * 32]; s += e[i]; }
#pragma unroll
  for (int m = 16; m >= 1; m >>= 1) s += __shfl_xor(s, m, 32);
  float mean = s * (1.0f / 512.0f);
  float v = 0.0f;
#pragma unroll
  for (int i = 0; i < 16; i++) { e[i] -= mean; v += e[i] * e[i]; }
#pragma unroll
  for (int m = 16; m >= 1; m >>= 1) v += __shfl_xor(v, m, 32);
  float rstd = rsqrtf(v * (1.0f / 512.0f) + 1e-5f);
  __bf16* y = Y + (size_t)row * DIM;
#pragma unroll
  for (int i = 0; i < 16; i++) {
    int c = lane + i * 32;
    y[c] = (__bf16)(e[i] * rstd * g[c] + bb[c]);
  }
}

// ----------------------------------------------------------------
extern "C" void kernel_launch(void* const* d_in, const int* in_sizes, int n_in,
                              void* d_out, int out_size, void* d_ws, size_t ws_size,
                              hipStream_t stream) {
  const float* x_q  = (const float*)d_in[0];
  const float* x_k  = (const float*)d_in[1];
  const float* x_v  = (const float*)d_in[2];
  const float* Wq   = (const float*)d_in[3];
  const float* Wk   = (const float*)d_in[4];
  const float* Wv   = (const float*)d_in[5];
  const float* Wp   = (const float*)d_in[6];
  const float* qn_g = (const float*)d_in[7];
  const float* qn_b = (const float*)d_in[8];
  const float* kn_g = (const float*)d_in[9];
  const float* kn_b = (const float*)d_in[10];
  const float* n_g  = (const float*)d_in[11];
  const float* n_b  = (const float*)d_in[12];

  const size_t TOK_E = (size_t)NTOK * DIM;
  const size_t W_E   = (size_t)DIM * DIM;
  char* ws = (char*)d_ws;
  size_t off = 0;
  auto take = [&](size_t bytes) { char* p = ws + off; off = (off + bytes + 255) & ~(size_t)255; return p; };

  __bf16* xq_bf = (__bf16*)take(TOK_E * 2);
  __bf16* xk_bf = (__bf16*)take(TOK_E * 2);
  __bf16* xv_bf = (__bf16*)take(TOK_E * 2);
  __bf16* wq_bf = (__bf16*)take(W_E * 2);
  __bf16* wk_bf = (__bf16*)take(W_E * 2);
  __bf16* wv_bf = (__bf16*)take(W_E * 2);
  __bf16* wp_bf = (__bf16*)take(W_E * 2);
  float*  qf    = (float*) take(TOK_E * 4);
  float*  kf    = (float*) take(TOK_E * 4);
  float*  vf    = (float*) take(TOK_E * 4);
  __bf16* qh_bf = (__bf16*)take(TOK_E * 2);
  __bf16* kh_bf = (__bf16*)take(TOK_E * 2);
  __bf16* vt_bf = (__bf16*)take(TOK_E * 2);
  float*  xattn = (float*) take(TOK_E * 4);
  __bf16* xln_bf= (__bf16*)take(TOK_E * 2);

  const int CB = 256;
  cvt_f32_bf16<<<(int)((TOK_E + CB - 1) / CB), CB, 0, stream>>>(x_q, xq_bf, (int)TOK_E);
  cvt_f32_bf16<<<(int)((TOK_E + CB - 1) / CB), CB, 0, stream>>>(x_k, xk_bf, (int)TOK_E);
  cvt_f32_bf16<<<(int)((TOK_E + CB - 1) / CB), CB, 0, stream>>>(x_v, xv_bf, (int)TOK_E);
  cvt_f32_bf16<<<(int)((W_E + CB - 1) / CB), CB, 0, stream>>>(Wq, wq_bf, (int)W_E);
  cvt_f32_bf16<<<(int)((W_E + CB - 1) / CB), CB, 0, stream>>>(Wk, wk_bf, (int)W_E);
  cvt_f32_bf16<<<(int)((W_E + CB - 1) / CB), CB, 0, stream>>>(Wv, wv_bf, (int)W_E);
  cvt_f32_bf16<<<(int)((W_E + CB - 1) / CB), CB, 0, stream>>>(Wp, wp_bf, (int)W_E);

  dim3 ggrid(NTOK / 128, DIM / 64);
  gemm_bf16_nt<<<ggrid, 256, 0, stream>>>(xq_bf, wq_bf, qf);
  gemm_bf16_nt<<<ggrid, 256, 0, stream>>>(xk_bf, wk_bf, kf);
  gemm_bf16_nt<<<ggrid, 256, 0, stream>>>(xv_bf, wv_bf, vf);

  headsplit<<<4096, 256, 0, stream>>>(qf, qh_bf, qn_g, qn_b, 0);
  headsplit<<<4096, 256, 0, stream>>>(kf, kh_bf, kn_g, kn_b, 1);
  headsplit<<<4096, 256, 0, stream>>>(vf, vt_bf, n_g, n_b, 2);  // g/b unused in mode 2

  attention<<<BATCH * NHEAD, 256, 0, stream>>>(qh_bf, kh_bf, vt_bf, xattn);

  row_ln512<<<NTOK / 8, 256, 0, stream>>>(xattn, n_g, n_b, xln_bf);
  gemm_bf16_nt<<<ggrid, 256, 0, stream>>>(xln_bf, wp_bf, (float*)d_out);
}
